// RSAFFDecoder_73358041416077
// MI455X (gfx1250) — matrix operand order repaired and checked by reading, compiled, not run on hardware
//
#include <hip/hip_runtime.h>
#include <hip/hip_bf16.h>

typedef __attribute__((ext_vector_type(16))) __bf16 v16bf;
typedef __attribute__((ext_vector_type(8)))  __bf16 v8bf;
typedef __attribute__((ext_vector_type(8)))  float  v8f;
typedef __attribute__((ext_vector_type(4))) unsigned int v4u;
typedef __attribute__((ext_vector_type(4))) int          v4i;
typedef __attribute__((ext_vector_type(8))) int          v8i;

namespace {
constexpr int kTPB = 256;
// Problem dims
constexpr int kHID = 512, kPROJ = 1536, kL = 4, kNC = 1000;
constexpr int kINP = 672; // 641 padded up to 21*32 for K-tiling

// Dynamic LDS layout (bytes)
constexpr int OFF_MEM   = 0;                      // f32  [32][512]   mem state
constexpr int OFF_MEMBF = OFF_MEM   + 32*512*4;   // bf16 [32][512]   mem (bf16 mirror / relu'd)
constexpr int OFF_BIG   = OFF_MEMBF + 32*512*2;   // bf16 [32][1536]  qkv / concat; also f32 [32][512] mlp-h
constexpr int OFF_STAGE = OFF_BIG   + 32*672*2 + 6144; // f32 [32][256] TDM staging (inside BIG tail)
constexpr int OFF_RED   = OFF_BIG   + 32*1536*2;  // f32  [512]       block reduce
constexpr int OFF_STAT  = OFF_RED   + 512*4;      // f32  [128]       LN stats
constexpr int OFF_D     = OFF_STAT  + 128*4;      // f32  [32]        softmaxed distances
constexpr int SMEM_BYTES = OFF_D + 32*4;          // ~195 KB (<= 320 KB per WGP)
static_assert(OFF_STAGE + 32*256*4 <= OFF_RED, "stage overlaps");
} // namespace

// ---- WMMA fragment loaders (CDNA5 16-bit layouts, wave32) ----
// A (16x32 bf16): lane<16 -> row=lane, K 0..7 & 16..23 ; lane>=16 -> row=lane-16, K 8..15 & 24..31
__device__ __forceinline__ v16bf load_a_frag(const __bf16* base, int lda,
                                             int tm, int k0, int lane) {
  const int row = tm * 16 + (lane & 15);
  const int kb  = k0 + ((lane >> 4) << 3);
  v8bf lo = *(const v8bf*)(base + row * lda + kb);
  v8bf hi = *(const v8bf*)(base + row * lda + kb + 16);
  return __builtin_shufflevector(lo, hi, 0,1,2,3,4,5,6,7,8,9,10,11,12,13,14,15);
}

// B (32x16 bf16) from row-major (out,in) weights: lane%16 = output col, K 0..7 & 16..23 (lane<16) / 8..15 & 24..31 (lane>=16)
__device__ __forceinline__ v16bf load_b_frag(const __bf16* base, int ldb,
                                             int tn, int k0, int lane) {
  const __bf16* p = base + (size_t)(tn * 16 + (lane & 15)) * ldb + k0 + ((lane >> 4) << 3);
  v8bf lo = *(const v8bf*)p;
  v8bf hi = *(const v8bf*)(p + 16);
  return __builtin_shufflevector(lo, hi, 0,1,2,3,4,5,6,7,8,9,10,11,12,13,14,15);
}

// One B fragment feeds BOTH row tiles (tm=0: rows 0..15, tm=1: rows 16..31):
// halves global weight traffic vs one-tile-per-wave.
__device__ __forceinline__ void wmma_gemm2(const __bf16* __restrict__ aLds, int lda,
                                           const __bf16* __restrict__ bGlb, int ldb,
                                           int tn, int ksteps, int lane,
                                           v8f& acc0, v8f& acc1) {
#pragma unroll 4
  for (int ks = 0; ks < ksteps; ++ks) {
    const int k0 = ks * 32;
    v16bf b  = load_b_frag(bGlb, ldb, tn, k0, lane);
    v16bf a0 = load_a_frag(aLds, lda, 0, k0, lane);
    v16bf a1 = load_a_frag(aLds, lda, 1, k0, lane);
    if (ks + 2 < ksteps) {
      __builtin_prefetch(bGlb + (size_t)(tn * 16 + (lane & 15)) * ldb + k0 + 64, 0, 3);
    }
    acc0 = __builtin_amdgcn_wmma_f32_16x16x32_bf16(false, a0, false, b,
                                                   (short)0, acc0, false, false);
    acc1 = __builtin_amdgcn_wmma_f32_16x16x32_bf16(false, a1, false, b,
                                                   (short)0, acc1, false, false);
  }
}

// ---- weight pre-conversion kernels (f32 -> bf16, once per launch) ----
__global__ void cvt_f32_bf16(const float* __restrict__ src, __bf16* __restrict__ dst, int n) {
  int i = blockIdx.x * blockDim.x + threadIdx.x;
  if (i < n) dst[i] = (__bf16)src[i];
}
__global__ void cvt_prew_pad(const float* __restrict__ src, __bf16* __restrict__ dst) {
  int i = blockIdx.x * blockDim.x + threadIdx.x;
  if (i >= 512 * kINP) return;
  int r = i / kINP, c = i - r * kINP;
  dst[i] = (c < 641) ? (__bf16)src[r * 641 + c] : (__bf16)0.0f;
}

// ---- main kernel: one workgroup per batch element ----
__global__ __launch_bounds__(kTPB)
void rsaff_main(const float* __restrict__ buf_emb,   // (B,32,256)
                const int*   __restrict__ labels,    // (B,32)
                const float* __restrict__ query,     // (B,256)
                const float* __restrict__ dists,     // (B,32)
                const float* __restrict__ emb_tab,   // (1001,128)
                const float* __restrict__ pre_b,     // (512)
                const float* __restrict__ qkv_b,     // (4,1536)
                const float* __restrict__ mlp_b,     // (4,512)
                const float* __restrict__ mlp_ln_w,  // (4,512)
                const float* __restrict__ mlp_ln_b,  // (4,512)
                const float* __restrict__ logit_b,   // (1000)
                const __bf16* __restrict__ w_pre,    // (512,672) padded
                const __bf16* __restrict__ w_qkv,    // (4,1536,512)
                const __bf16* __restrict__ lnw_qkv,  // (4,32,1536)
                const __bf16* __restrict__ lnb_qkv,  // (4,32,1536)
                const __bf16* __restrict__ w_mlp,    // (4,512,512)
                const __bf16* __restrict__ w_logit,  // (1000,512)
                float* __restrict__ out)             // (B,1000)
{
  extern __shared__ char smem[];
  float*  s_mem   = (float*)(smem + OFF_MEM);
  __bf16* s_membf = (__bf16*)(smem + OFF_MEMBF);
  __bf16* s_big   = (__bf16*)(smem + OFF_BIG);
  float*  s_hf    = (float*)(smem + OFF_BIG);    // aliases s_big (phase-disjoint)
  float*  s_stage = (float*)(smem + OFF_STAGE);  // TDM staging for buffer_embeds
  float*  s_red   = (float*)(smem + OFF_RED);
  float*  s_stat  = (float*)(smem + OFF_STAT);
  float*  s_d     = (float*)(smem + OFF_D);

  const int t    = threadIdx.x;
  const int lane = t & 31;
  const int wv   = t >> 5;
  const int b    = blockIdx.x;

  // ---- kick off TDM: DMA this block's 32KB buffer_embeds slab into LDS ----
  if (wv == 0) {
    const unsigned lds_byte = (unsigned)(size_t)s_stage; // low 32 bits = LDS offset
    const unsigned long long ga =
        (unsigned long long)(size_t)(buf_emb + (size_t)b * 32 * 256);
    // D# group0: count=1, lds_addr, global_addr[56:0], type=2 ("image")
    v4u g0 = { 1u,
               lds_byte,
               (unsigned)(ga & 0xFFFFFFFFu),
               (unsigned)((ga >> 32) & 0x01FFFFFFu) | (2u << 30) };
    // D# group1: data_size=4B; 1-D tensor/tile of 8192 elements
    const unsigned td0 = 8192, tile0 = 8192, td1 = 1;
    v8i g1 = { (int)(2u << 16),                                    // data_size=2 -> 4B
               (int)((td0 & 0xFFFFu) << 16),                       // tensor_dim0 lo16
               (int)((td0 >> 16) | ((td1 & 0xFFFFu) << 16)),       // dim0 hi | dim1 lo
               (int)((td1 >> 16) | (tile0 << 16)),                 // dim1 hi | tile_dim0
               0,                                                  // tile_dim1/2 unused
               (int)td0,                                           // dim0_stride lo32
               0, 0 };
    v4i gz = { 0, 0, 0, 0 };
#if defined(__clang_major__) && (__clang_major__ >= 23)
    v8i gz8 = { 0, 0, 0, 0, 0, 0, 0, 0 };
    __builtin_amdgcn_tensor_load_to_lds(g0, g1, gz, gz, gz8, 0);
#else
    __builtin_amdgcn_tensor_load_to_lds(g0, g1, gz, gz, 0);
#endif
  }

  // ---- d = softmax(-distances[b])  (overlaps the TDM transfer) ----
  if (t < 32) s_d[t] = -dists[(size_t)b * 32 + t];
  __syncthreads();
  if (t == 0) {
    float mx = s_d[0];
    for (int i = 1; i < 32; ++i) mx = fmaxf(mx, s_d[i]);
    float sm = 0.f;
    for (int i = 0; i < 32; ++i) sm += __expf(s_d[i] - mx);
    s_stat[0] = mx; s_stat[1] = sm;
  }
  __syncthreads();
  if (t < 32) s_d[t] = __expf(s_d[t] - s_stat[0]) / s_stat[1];
  if (wv == 0) __builtin_amdgcn_s_wait_tensorcnt(0);  // TDM slab landed
  __syncthreads();

  // ---- build concat row (bf16, K padded to 672) into s_big ----
  for (int e = t; e < 32 * kINP; e += kTPB) {
    int n = e / kINP, c = e - n * kINP;
    float v;
    if (c < 256)       v = s_stage[n * 256 + c];                 // from TDM staging
    else if (c < 384)  v = emb_tab[(size_t)labels[(size_t)b * 32 + n] * 128 + (c - 256)];
    else if (c < 640)  v = query[(size_t)b * 256 + (c - 384)];
    else if (c == 640) v = s_d[n];
    else               v = 0.f;
    s_big[e] = (__bf16)v;
  }
  __syncthreads();

  // ---- mem = concat @ pre_w^T + pre_b  (32 col-tiles, both row-tiles per wave) ----
  for (int tn = wv; tn < 32; tn += 8) {
    v8f acc0 = {}, acc1 = {};
    wmma_gemm2(s_big, kINP, w_pre, kINP, tn, 21, lane, acc0, acc1);
#pragma unroll
    for (int v = 0; v < 8; ++v) {
      int r0  = v + ((lane >> 4) << 3);
      int col = tn * 16 + (lane & 15);
      float bias = pre_b[col];
      s_mem[r0 * kHID + col]        = acc0[v] + bias;
      s_mem[(r0 + 16) * kHID + col] = acc1[v] + bias;
    }
  }
  __syncthreads();
  for (int e = t; e < 32 * kHID; e += kTPB) s_membf[e] = (__bf16)s_mem[e];
  __syncthreads();

  // ---- transformer layers ----
  for (int l = 0; l < kL; ++l) {
    // qkv = mem @ qkv_w^T + qkv_b    (96 col-tiles, K=16*32)
    const __bf16* wq = w_qkv + (size_t)l * kPROJ * kHID;
    for (int tn = wv; tn < 96; tn += 8) {
      v8f acc0 = {}, acc1 = {};
      wmma_gemm2(s_membf, kHID, wq, kHID, tn, 16, lane, acc0, acc1);
#pragma unroll
      for (int v = 0; v < 8; ++v) {
        int r0  = v + ((lane >> 4) << 3);
        int col = tn * 16 + (lane & 15);
        float bias = qkv_b[l * kPROJ + col];
        s_big[r0 * kPROJ + col]        = (__bf16)(acc0[v] + bias);
        s_big[(r0 + 16) * kPROJ + col] = (__bf16)(acc1[v] + bias);
      }
    }
    __syncthreads();

    // LayerNorm over all (N,PROJ): block reduce sum/sumsq
    float ps = 0.f, pq = 0.f;
    for (int e = t; e < 32 * kPROJ; e += kTPB) {
      float x = (float)s_big[e];
      ps += x; pq += x * x;
    }
    s_red[t] = ps; s_red[256 + t] = pq;
    __syncthreads();
    for (int st = 128; st > 0; st >>= 1) {
      if (t < st) { s_red[t] += s_red[t + st]; s_red[256 + t] += s_red[256 + t + st]; }
      __syncthreads();
    }
    if (t == 0) {
      float mu  = s_red[0] / 49152.f;
      float var = s_red[256] / 49152.f - mu * mu;
      s_stat[0] = mu; s_stat[1] = rsqrtf(var + 1e-5f);
    }
    __syncthreads();
    {
      const float mu = s_stat[0], rstd = s_stat[1];
      const __bf16* lw = lnw_qkv + (size_t)l * 49152;
      const __bf16* lb = lnb_qkv + (size_t)l * 49152;
      for (int e = t; e < 49152; e += kTPB) {
        float x = (float)s_big[e];
        s_big[e] = (__bf16)((x - mu) * rstd * (float)lw[e] + (float)lb[e]);
      }
    }
    __syncthreads();

    // Attention (16 heads, N=32, VALU f32) + residual into s_mem
    for (int task = t; task < 512; task += kTPB) {
      int h = task >> 5, n = task & 31;
      const __bf16* base = s_big + h * 96;
      float qv[32], p[32];
#pragma unroll
      for (int c = 0; c < 32; ++c)
        qv[c] = (float)base[n * kPROJ + c] * 0.17677669529663687f; // KS^-0.5
      float mx = -1e30f;
      for (int m = 0; m < 32; ++m) {
        const __bf16* kr = base + m * kPROJ + 32;
        float dot = 0.f;
#pragma unroll
        for (int c = 0; c < 32; ++c) dot += qv[c] * (float)kr[c];
        p[m] = dot; mx = fmaxf(mx, dot);
      }
      float sm = 0.f;
      for (int m = 0; m < 32; ++m) { p[m] = __expf(p[m] - mx); sm += p[m]; }
      float inv = 1.f / sm;
      for (int c = 0; c < 32; ++c) {
        float o = 0.f;
#pragma unroll
        for (int m = 0; m < 32; ++m) o += p[m] * (float)base[m * kPROJ + 64 + c];
        s_mem[n * kHID + h * 32 + c] += o * inv;
      }
    }
    __syncthreads();

    // s_membf = bf16(relu(mem)) for MLP
    for (int e = t; e < 32 * kHID; e += kTPB) s_membf[e] = (__bf16)fmaxf(s_mem[e], 0.f);
    __syncthreads();

    // h = relu(mem) @ mlp_w^T + mlp_b   (32 col-tiles, f32 into s_hf)
    const __bf16* wm = w_mlp + (size_t)l * kHID * kHID;
    for (int tn = wv; tn < 32; tn += 8) {
      v8f acc0 = {}, acc1 = {};
      wmma_gemm2(s_membf, kHID, wm, kHID, tn, 16, lane, acc0, acc1);
#pragma unroll
      for (int v = 0; v < 8; ++v) {
        int r0  = v + ((lane >> 4) << 3);
        int col = tn * 16 + (lane & 15);
        float bias = mlp_b[l * kHID + col];
        s_hf[r0 * kHID + col]        = acc0[v] + bias;
        s_hf[(r0 + 16) * kHID + col] = acc1[v] + bias;
      }
    }
    __syncthreads();

    // per-row LN(h) + residual into mem
    {
      int row = t >> 3, seg = t & 7;
      const float* hp = s_hf + row * kHID + seg * 64;
      float ps2 = 0.f, pq2 = 0.f;
      for (int i = 0; i < 64; ++i) { float x = hp[i]; ps2 += x; pq2 += x * x; }
      s_red[t] = ps2; s_red[256 + t] = pq2;
      __syncthreads();
      if (seg == 0) {
        float sm2 = 0.f, sq2 = 0.f;
        for (int i = 0; i < 8; ++i) { sm2 += s_red[row * 8 + i]; sq2 += s_red[256 + row * 8 + i]; }
        float mu2  = sm2 / 512.f;
        float var2 = sq2 / 512.f - mu2 * mu2;
        s_stat[2 + row * 2] = mu2;
        s_stat[3 + row * 2] = rsqrtf(var2 + 1e-5f);
      }
      __syncthreads();
      for (int e = t; e < 32 * kHID; e += kTPB) {
        int n = e >> 9, c = e & 511;
        float mu2 = s_stat[2 + n * 2], rs2 = s_stat[3 + n * 2];
        s_mem[e] += (s_hf[e] - mu2) * rs2 * mlp_ln_w[l * kHID + c] + mlp_ln_b[l * kHID + c];
      }
    }
    __syncthreads();

    // refresh bf16 mirror for next layer's qkv GEMM
    for (int e = t; e < 32 * kHID; e += kTPB) s_membf[e] = (__bf16)s_mem[e];
    __syncthreads();
  }

  // ---- weighted = sum_n mem[n]*d[n] (bf16 into s_membf[0..511]) ----
  for (int c = t; c < kHID; c += kTPB) {
    float acc = 0.f;
#pragma unroll
    for (int n = 0; n < 32; ++n) acc += s_mem[n * kHID + c] * s_d[n];
    s_membf[c] = (__bf16)acc;
  }
  __syncthreads();

  // ---- logits = weighted @ logit_w^T + logit_b ----
  for (int c = t; c < kNC; c += kTPB) {
    const __bf16* wr = w_logit + (size_t)c * kHID;
    float acc = 0.f;
    for (int i = 0; i < kHID; i += 8) {
      v8bf w8 = *(const v8bf*)(wr + i);
      v8bf x8 = *(const v8bf*)(s_membf + i);
#pragma unroll
      for (int j = 0; j < 8; ++j) acc += (float)x8[j] * (float)w8[j];
    }
    out[(size_t)b * kNC + c] = acc + logit_b[c];
  }
}

extern "C" void kernel_launch(void* const* d_in, const int* in_sizes, int n_in,
                              void* d_out, int out_size, void* d_ws, size_t ws_size,
                              hipStream_t stream) {
  const float* buf_emb   = (const float*)d_in[0];
  const int*   labels    = (const int*)  d_in[1];
  const float* query     = (const float*)d_in[2];
  const float* dists     = (const float*)d_in[3];
  const float* emb_tab   = (const float*)d_in[4];
  const float* pre_w     = (const float*)d_in[5];
  const float* pre_b     = (const float*)d_in[6];
  const float* qkv_w     = (const float*)d_in[7];
  const float* qkv_b     = (const float*)d_in[8];
  const float* qkv_ln_w  = (const float*)d_in[9];
  const float* qkv_ln_b  = (const float*)d_in[10];
  const float* mlp_w     = (const float*)d_in[11];
  const float* mlp_b     = (const float*)d_in[12];
  const float* mlp_ln_w  = (const float*)d_in[13];
  const float* mlp_ln_b  = (const float*)d_in[14];
  const float* logit_w   = (const float*)d_in[15];
  const float* logit_b   = (const float*)d_in[16];
  (void)in_sizes; (void)n_in; (void)out_size; (void)ws_size;

  // carve bf16 weight copies out of workspace (~11 MB)
  char* ws = (char*)d_ws;
  size_t off = 0;
  auto carve = [&](size_t bytes) { void* p = ws + off; off += (bytes + 255) & ~(size_t)255; return p; };
  __bf16* w_pre   = (__bf16*)carve((size_t)512 * kINP * 2);
  __bf16* w_qkv   = (__bf16*)carve((size_t)kL * kPROJ * kHID * 2);
  __bf16* lnw     = (__bf16*)carve((size_t)kL * 32 * kPROJ * 2);
  __bf16* lnb     = (__bf16*)carve((size_t)kL * 32 * kPROJ * 2);
  __bf16* w_mlp   = (__bf16*)carve((size_t)kL * kHID * kHID * 2);
  __bf16* w_logit = (__bf16*)carve((size_t)kNC * kHID * 2);

  auto cvt = [&](const float* s, __bf16* d, int n) {
    cvt_f32_bf16<<<(n + 255) / 256, 256, 0, stream>>>(s, d, n);
  };
  cvt_prew_pad<<<(512 * kINP + 255) / 256, 256, 0, stream>>>(pre_w, w_pre);
  cvt(qkv_w,    w_qkv,   kL * kPROJ * kHID);
  cvt(qkv_ln_w, lnw,     kL * 32 * kPROJ);
  cvt(qkv_ln_b, lnb,     kL * 32 * kPROJ);
  cvt(mlp_w,    w_mlp,   kL * kHID * kHID);
  cvt(logit_w,  w_logit, kNC * kHID);

  rsaff_main<<<4096, kTPB, SMEM_BYTES, stream>>>(
      buf_emb, labels, query, dists, emb_tab,
      pre_b, qkv_b, mlp_b, mlp_ln_w, mlp_ln_b, logit_b,
      w_pre, w_qkv, lnw, lnb, w_mlp, w_logit, (float*)d_out);
}
